// MPNNModel_28664611733514
// MI455X (gfx1250) — matrix operand order repaired and checked by reading, not compiled, not run on hardware
//
#include <hip/hip_runtime.h>
#include <cstdint>

// ---------------------------------------------------------------------------
// MPNN on MI455X (gfx1250, wave32). All dense MLPs run on
// v_wmma_f32_16x16x32_f16 with activations in f16 and f32 accumulation.
// Weights are pre-packed into WMMA-B fragment layout (ISA 7.12.2):
//   tile(kt,nt): lane l, half i  ->  W[kt*32 + ((l>=16)?8:0) + (i&7) + ((i&8)?16:0)][nt*16 + (l&15)]
// stored contiguously (32 lanes * 16 halves = 1KB per tile) for coalesced loads.
// A-fragments are read from LDS tiles per the 16-bit A layout:
//   lane<16 : M=lane,    K = {0..7, 16..23}   (+kt*32)
//   lane>=16: M=lane-16, K = {8..15, 24..31}  (+kt*32)
// C/D layout: VGPR r holds M = r + (lane>=16 ? 8 : 0), N = lane&15.
// ---------------------------------------------------------------------------

#define NNODES 100000
#define NEDGES 600000
#define BGRAPH 512
#define EW 2   // waves per block, edge/node kernels (keeps static LDS < 64KB)
#define KW 4   // waves per block, kv kernel

typedef __attribute__((ext_vector_type(16))) _Float16 v16h;
typedef __attribute__((ext_vector_type(8)))  float    v8f;

__device__ __forceinline__ v16h ldsA_frag(const _Float16* A, int lda, int kt, int lane) {
  const _Float16* p = A + (size_t)(lane & 15) * lda + kt * 32 + ((lane & 16) ? 8 : 0);
  v16h a;
#pragma unroll
  for (int i = 0; i < 8; ++i) { a[i] = p[i]; a[i + 8] = p[i + 16]; }
  return a;
}

__device__ __forceinline__ v16h ldB_frag(const _Float16* __restrict__ Bp, int lane) {
  const _Float16* p = Bp + lane * 16;   // 32 bytes contiguous per lane
  v16h b;
#pragma unroll
  for (int i = 0; i < 16; ++i) b[i] = p[i];
  return b;
}

// LDS[16xK] (stride lda) x packedB[KT][NT][512] + bias  -> LDS[16xN] (stride ldo)
__device__ __forceinline__ void gemm_lds(const _Float16* A, int lda,
                                         const _Float16* __restrict__ Bp,
                                         const float* __restrict__ bias,
                                         int KT, int NT,
                                         _Float16* O, int ldo, bool relu, int lane) {
  const int mb  = (lane & 16) ? 8 : 0;
  const int nlo = lane & 15;
  for (int nt = 0; nt < NT; ++nt) {
    float bv = bias ? bias[nt * 16 + nlo] : 0.0f;
    v8f acc = {bv, bv, bv, bv, bv, bv, bv, bv};
    for (int kt = 0; kt < KT; ++kt) {
      v16h a = ldsA_frag(A, lda, kt, lane);
      v16h b = ldB_frag(Bp + (size_t)(kt * NT + nt) * 512, lane);
      acc = __builtin_amdgcn_wmma_f32_16x16x32_f16(false, a, false, b, (short)0, acc,
                                                   false, false);
    }
    int col = nt * 16 + nlo;
#pragma unroll
    for (int r = 0; r < 8; ++r) {
      float x = acc[r];
      if (relu) x = fmaxf(x, 0.0f);
      O[(size_t)(mb + r) * ldo + col] = (_Float16)x;
    }
  }
}

// ------------------------------- utility kernels ---------------------------

__global__ void k_fill_u32(unsigned* __restrict__ p, unsigned v, long n) {
  long t = (long)blockIdx.x * blockDim.x + threadIdx.x;
  if (t < n) p[t] = v;
}

// Pack f32 row-major weight [K,N] into f16 WMMA-B fragment layout.
__global__ void k_pack_w(const float* __restrict__ W, int K, int N,
                         _Float16* __restrict__ out) {
  int t = blockIdx.x * blockDim.x + threadIdx.x;
  int total = K * N;
  if (t >= total) return;
  int NT = N >> 4;
  int tile = t >> 9, rem = t & 511, lane = rem >> 4, i = rem & 15;
  int kt = tile / NT, nt = tile % NT;
  int k = kt * 32 + ((lane & 16) ? 8 : 0) + (i & 7) + ((i & 8) ? 16 : 0);
  int n = nt * 16 + (lane & 15);
  out[t] = (_Float16)W[(size_t)k * N + n];
}

// Pack seeds [S=4,H=4,dh=32] into a block-diagonal 128x16 WMMA-B matrix:
// column col = h*4+s ; B[k][col] = (k>>5 == h) ? seeds[s][h][k&31] : 0
__global__ void k_pack_seeds(const float* __restrict__ seeds, _Float16* __restrict__ out) {
  int t = blockIdx.x * blockDim.x + threadIdx.x;
  if (t >= 2048) return;
  int tile = t >> 9, rem = t & 511, lane = rem >> 4, i = rem & 15;
  int k = tile * 32 + ((lane & 16) ? 8 : 0) + (i & 7) + ((i & 8) ? 16 : 0);   // 0..127
  int n = lane & 15;
  int hh = k >> 5, d = k & 31, ch = n >> 2, cs = n & 3;
  float v = (hh == ch) ? seeds[(cs * 4 + hh) * 32 + d] : 0.0f;
  out[t] = (_Float16)v;
}

// ------------------------------- input projections -------------------------

__global__ __launch_bounds__(128) void k_node_in(
    const int* __restrict__ atype, const float* __restrict__ xfeats,
    const float* __restrict__ aemb, const float* __restrict__ w,
    const float* __restrict__ b, _Float16* __restrict__ h) {
  __shared__ float xin[72];
  int n = blockIdx.x, t = threadIdx.x;
  if (t < 64)      xin[t] = aemb[(size_t)atype[n] * 64 + t];
  else if (t < 72) xin[t] = xfeats[(size_t)n * 8 + (t - 64)];
  __syncthreads();
  float acc = b[t];
#pragma unroll 8
  for (int i = 0; i < 72; ++i) acc += xin[i] * w[i * 128 + t];
  h[(size_t)n * 128 + t] = (_Float16)acc;
}

__global__ __launch_bounds__(128) void k_edge_in(
    const int* __restrict__ btype, const float* __restrict__ efeats,
    const float* __restrict__ bemb, const float* __restrict__ w,
    const float* __restrict__ b, _Float16* __restrict__ me) {
  __shared__ float xin[19];
  int e = blockIdx.x, t = threadIdx.x;
  if (t < 16)       xin[t]  = bemb[(size_t)btype[e] * 16 + t];
  else if (t == 16) xin[16] = efeats[(size_t)e * 4 + 0];
  else if (t == 17) xin[17] = efeats[(size_t)e * 4 + 2];
  else if (t == 18) xin[18] = efeats[(size_t)e * 4 + 3];
  __syncthreads();
  float acc = b[t];
#pragma unroll
  for (int i = 0; i < 19; ++i) acc += xin[i] * w[i * 128 + t];
  me[(size_t)e * 128 + t] = (_Float16)acc;
}

// ------------------------------- MP layer: edges ---------------------------
// One wave handles a 16-edge tile end-to-end (no inter-wave sync needed).

__global__ __launch_bounds__(64) void k_edge_layer(
    const _Float16* __restrict__ h, _Float16* __restrict__ msg_e,
    float* __restrict__ aggr,
    const int* __restrict__ src, const int* __restrict__ dst,
    const _Float16* __restrict__ m1w, const float* __restrict__ m1b,
    const _Float16* __restrict__ m2w, const float* __restrict__ m2b,
    const _Float16* __restrict__ e1w, const float* __restrict__ e1b,
    const _Float16* __restrict__ e2w, const float* __restrict__ e2b,
    const _Float16* __restrict__ e3w, const float* __restrict__ e3b,
    int ntiles) {
  __shared__ _Float16 zb[EW][16][392];   // z = [h_src | h_dst | msg_e], padded
  __shared__ _Float16 tb[EW][16][264];   // 256-wide hidden, padded
  __shared__ _Float16 ub[EW][16][136];   // 128-wide hidden, padded
  __shared__ int sidx[EW][32];           // [0..15]=src, [16..31]=dst
  const int wv = threadIdx.x >> 5, lane = threadIdx.x & 31;
  const int tile = blockIdx.x * EW + wv;
  if (tile >= ntiles) return;
  const int e0 = tile * 16;

  __builtin_prefetch(m1w + (size_t)lane * 256, 0, 3);   // warm weight lines in L2
  __builtin_prefetch(e1w + (size_t)lane * 256, 0, 3);

  if (lane < 16) {
    sidx[wv][lane]      = src[e0 + lane];
    sidx[wv][16 + lane] = dst[e0 + lane];
  }
  // stage z tile: 16 rows x 384 f16 (gathered)
  for (int r = 0; r < 16; ++r) {
    const unsigned* ps = (const unsigned*)(h + (size_t)sidx[wv][r] * 128);
    const unsigned* pd = (const unsigned*)(h + (size_t)sidx[wv][16 + r] * 128);
    const unsigned* pe = (const unsigned*)(msg_e + (size_t)(e0 + r) * 128);
    unsigned* zr = (unsigned*)&zb[wv][r][0];
    zr[lane]       = ps[lane];       zr[32 + lane]  = ps[32 + lane];
    zr[64 + lane]  = pd[lane];       zr[96 + lane]  = pd[32 + lane];
    zr[128 + lane] = pe[lane];       zr[160 + lane] = pe[32 + lane];
  }
  const int mb = (lane & 16) ? 8 : 0, nlo = lane & 15;

  // ---- message path: relu(z @ m1) @ m2, scatter-add into aggr[dst]
  gemm_lds(&zb[wv][0][0], 392, m1w, m1b, 12, 16, &tb[wv][0][0], 264, true, lane);
  for (int nt = 0; nt < 8; ++nt) {
    float bv = m2b[nt * 16 + nlo];
    v8f acc = {bv, bv, bv, bv, bv, bv, bv, bv};
    for (int kt = 0; kt < 8; ++kt) {
      v16h a = ldsA_frag(&tb[wv][0][0], 264, kt, lane);
      v16h b = ldB_frag(m2w + (size_t)(kt * 8 + nt) * 512, lane);
      acc = __builtin_amdgcn_wmma_f32_16x16x32_f16(false, a, false, b, (short)0, acc,
                                                   false, false);
    }
    int col = nt * 16 + nlo;
#pragma unroll
    for (int r = 0; r < 8; ++r) {
      int d = sidx[wv][16 + mb + r];
      atomicAdd(&aggr[(size_t)d * 128 + col], acc[r]);
    }
  }

  // ---- edge path: relu(relu(z@e1)@e2)@e3, residual into msg_e
  gemm_lds(&zb[wv][0][0], 392, e1w, e1b, 12, 16, &tb[wv][0][0], 264, true, lane);
  gemm_lds(&tb[wv][0][0], 264, e2w, e2b, 8, 8, &ub[wv][0][0], 136, true, lane);
  for (int nt = 0; nt < 8; ++nt) {
    float bv = e3b[nt * 16 + nlo];
    v8f acc = {bv, bv, bv, bv, bv, bv, bv, bv};
    for (int kt = 0; kt < 4; ++kt) {
      v16h a = ldsA_frag(&ub[wv][0][0], 136, kt, lane);
      v16h b = ldB_frag(e3w + (size_t)(kt * 8 + nt) * 512, lane);
      acc = __builtin_amdgcn_wmma_f32_16x16x32_f16(false, a, false, b, (short)0, acc,
                                                   false, false);
    }
    int col = nt * 16 + nlo;
#pragma unroll
    for (int r = 0; r < 8; ++r) {
      float x = acc[r] + (float)zb[wv][mb + r][256 + col];   // + old msg_e
      msg_e[(size_t)(e0 + mb + r) * 128 + col] = (_Float16)x;
    }
  }
}

// ------------------------------- MP layer: nodes ---------------------------

__global__ __launch_bounds__(64) void k_node_layer(
    _Float16* __restrict__ h, float* __restrict__ aggr,
    const _Float16* __restrict__ u1w, const float* __restrict__ u1b,
    const _Float16* __restrict__ u2w, const float* __restrict__ u2b,
    int ntiles) {
  __shared__ _Float16 ab[EW][16][264];   // [h | aggr]
  __shared__ _Float16 tb[EW][16][264];
  const int wv = threadIdx.x >> 5, lane = threadIdx.x & 31;
  const int tile = blockIdx.x * EW + wv;
  if (tile >= ntiles) return;
  const int n0 = tile * 16;
  for (int r = 0; r < 16; ++r) {
    const unsigned* ph = (const unsigned*)(h + (size_t)(n0 + r) * 128);
    unsigned* ar = (unsigned*)&ab[wv][r][0];
    ar[lane] = ph[lane]; ar[32 + lane] = ph[32 + lane];
    float* ag = aggr + (size_t)(n0 + r) * 128;
#pragma unroll
    for (int j = 0; j < 4; ++j) {
      int c = lane * 4 + j;
      ab[wv][r][128 + c] = (_Float16)ag[c];
      ag[c] = 0.0f;                       // re-zero aggr for next layer
    }
  }
  gemm_lds(&ab[wv][0][0], 264, u1w, u1b, 8, 16, &tb[wv][0][0], 264, true, lane);
  const int mb = (lane & 16) ? 8 : 0, nlo = lane & 15;
  for (int nt = 0; nt < 8; ++nt) {
    float bv = u2b[nt * 16 + nlo];
    v8f acc = {bv, bv, bv, bv, bv, bv, bv, bv};
    for (int kt = 0; kt < 8; ++kt) {
      v16h a = ldsA_frag(&tb[wv][0][0], 264, kt, lane);
      v16h b = ldB_frag(u2w + (size_t)(kt * 8 + nt) * 512, lane);
      acc = __builtin_amdgcn_wmma_f32_16x16x32_f16(false, a, false, b, (short)0, acc,
                                                   false, false);
    }
    int col = nt * 16 + nlo;
#pragma unroll
    for (int r = 0; r < 8; ++r) {
      size_t off = (size_t)(n0 + mb + r) * 128 + col;
      h[off] = (_Float16)((float)ab[wv][mb + r][col] + acc[r]);   // residual
    }
  }
}

// ------------------------------- attention readout -------------------------

__global__ __launch_bounds__(128) void k_kv(
    const _Float16* __restrict__ h,
    const _Float16* __restrict__ kw, const float* __restrict__ kb,
    const _Float16* __restrict__ vw, const float* __restrict__ vb,
    const _Float16* __restrict__ sB,
    const int* __restrict__ batch,
    float* __restrict__ scores, unsigned* __restrict__ smax,
    _Float16* __restrict__ vbuf, int ntiles) {
  __shared__ _Float16 hb[KW][16][136];
  __shared__ _Float16 kbuf[KW][16][136];
  __shared__ int sbt[KW][16];
  const int wv = threadIdx.x >> 5, lane = threadIdx.x & 31;
  const int tile = blockIdx.x * KW + wv;
  if (tile >= ntiles) return;
  const int n0 = tile * 16;
  if (lane < 16) sbt[wv][lane] = batch[n0 + lane];
  for (int r = 0; r < 16; ++r) {
    const unsigned* ph = (const unsigned*)(h + (size_t)(n0 + r) * 128);
    unsigned* hr = (unsigned*)&hb[wv][r][0];
    hr[lane] = ph[lane]; hr[32 + lane] = ph[32 + lane];
  }
  // K projection
  gemm_lds(&hb[wv][0][0], 136, kw, kb, 4, 8, &kbuf[wv][0][0], 136, false, lane);
  // scores = K @ seedsB (block-diagonal 128x16), col = h*4+s
  v8f acc = {0.f, 0.f, 0.f, 0.f, 0.f, 0.f, 0.f, 0.f};
  for (int kt = 0; kt < 4; ++kt) {
    v16h a = ldsA_frag(&kbuf[wv][0][0], 136, kt, lane);
    v16h b = ldB_frag(sB + (size_t)kt * 512, lane);
    acc = __builtin_amdgcn_wmma_f32_16x16x32_f16(false, a, false, b, (short)0, acc,
                                                 false, false);
  }
  const int mb = (lane & 16) ? 8 : 0, nlo = lane & 15;
  const float scl = 0.17677669529663687f;   // 1/sqrt(32)
#pragma unroll
  for (int r = 0; r < 8; ++r) {
    int n = n0 + mb + r;
    float s = acc[r] * scl;
    scores[(size_t)n * 16 + nlo] = s;
    unsigned u = __float_as_uint(s);
    unsigned enc = (u & 0x80000000u) ? ~u : (u | 0x80000000u);   // order-preserving
    atomicMax(&smax[sbt[wv][mb + r] * 16 + nlo], enc);
  }
  // V projection -> global f16
  gemm_lds(&hb[wv][0][0], 136, vw, vb, 4, 8, &kbuf[wv][0][0], 136, false, lane);
  for (int r = 0; r < 16; ++r) {
    unsigned* pv = (unsigned*)(vbuf + (size_t)(n0 + r) * 128);
    const unsigned* kr = (const unsigned*)&kbuf[wv][r][0];
    pv[lane] = kr[lane]; pv[32 + lane] = kr[32 + lane];
  }
}

__global__ void k_expsum(float* __restrict__ scores, const unsigned* __restrict__ smax,
                         float* __restrict__ denom, const int* __restrict__ batch,
                         int nNodes) {
  int t = blockIdx.x * blockDim.x + threadIdx.x;
  if (t >= nNodes * 16) return;
  int node = t >> 4, c = t & 15, b = batch[node];
  unsigned enc = smax[b * 16 + c];
  float mx = (enc & 0x80000000u) ? __uint_as_float(enc ^ 0x80000000u)
                                 : __uint_as_float(~enc);
  float ex = __expf(scores[t] - mx);
  scores[t] = ex;
  atomicAdd(&denom[b * 16 + c], ex);
}

__global__ __launch_bounds__(128) void k_pooled(
    const float* __restrict__ scores, const float* __restrict__ denom,
    const _Float16* __restrict__ vbuf, const int* __restrict__ batch,
    float* __restrict__ pooled) {
  int n = blockIdx.x, b = batch[n];
#pragma unroll
  for (int j = 0; j < 4; ++j) {
    int idx = threadIdx.x * 4 + j;            // (h*4+s)*32 + d
    int col = idx >> 5, d = idx & 31, head = col >> 2;
    float attn = scores[(size_t)n * 16 + col] / denom[b * 16 + col];
    float v = (float)vbuf[(size_t)n * 128 + head * 32 + d];
    atomicAdd(&pooled[(size_t)b * 512 + idx], attn * v);
  }
}

__global__ __launch_bounds__(128) void k_epool(
    const _Float16* __restrict__ msg_e, const int* __restrict__ src,
    const int* __restrict__ batch, float* __restrict__ esum,
    float* __restrict__ ecnt) {
  int e = blockIdx.x, t = threadIdx.x;
  int b = batch[src[e]];
  atomicAdd(&esum[(size_t)b * 128 + t], (float)msg_e[(size_t)e * 128 + t]);
  if (t == 0) atomicAdd(&ecnt[b], 1.0f);
}

// ------------------------------- final MLP ---------------------------------

__global__ __launch_bounds__(32) void k_final(
    const float* __restrict__ pooled, const float* __restrict__ esum,
    const float* __restrict__ ecnt,
    const float* __restrict__ dielec, const float* __restrict__ refract,
    const float* __restrict__ dw, const float* __restrict__ db,
    const float* __restrict__ rw, const float* __restrict__ rb,
    const _Float16* __restrict__ o1w, const float* __restrict__ o1b,
    const float* __restrict__ o2w, const float* __restrict__ o2b,
    float* __restrict__ out) {
  __shared__ _Float16 zb[16][712];   // 704 + pad
  __shared__ _Float16 tb[16][264];
  const int lane = threadIdx.x;
  const int g0 = blockIdx.x * 16;
  for (int r = 0; r < 16; ++r) {
    int g = g0 + r;
#pragma unroll
    for (int j = 0; j < 16; ++j)
      zb[r][lane + 32 * j] = (_Float16)pooled[(size_t)g * 512 + lane + 32 * j];
    float inv = 1.0f / fmaxf(ecnt[g], 1.0f);
#pragma unroll
    for (int j = 0; j < 4; ++j) {
      int c = lane + 32 * j;
      zb[r][512 + c] = (_Float16)(esum[(size_t)g * 128 + c] * inv);
    }
    zb[r][640 + lane] = (_Float16)fmaxf(dielec[g] * dw[lane] + db[lane], 0.0f);
    zb[r][672 + lane] = (_Float16)fmaxf(refract[g] * rw[lane] + rb[lane], 0.0f);
  }
  gemm_lds(&zb[0][0], 712, o1w, o1b, 22, 16, &tb[0][0], 264, true, lane);
  for (int r = 0; r < 16; ++r) {
    float s = 0.0f;
#pragma unroll
    for (int i = 0; i < 8; ++i)
      s += (float)tb[r][lane * 8 + i] * o2w[lane * 8 + i];
    for (int off = 16; off; off >>= 1) s += __shfl_xor(s, off, 32);
    if (lane == 0) out[g0 + r] = s + o2b[0];
  }
}

// ------------------------------- host driver -------------------------------
// d_in layout: top-level dict insertion order, then `params` flattened as a
// jax pytree (keys sorted): atom_emb, bond_emb, dielec{b,w}, k{b,w},
// layers[0..3]{edge1..3,msg1..2,upd1..2 each b,w}, lin_in_atoms{b,w},
// lin_in_bonds{b,w}, out1{b,w}, out2{b,w}, refract{b,w}, seeds, v{b,w}.

extern "C" void kernel_launch(void* const* d_in, const int* in_sizes, int n_in,
                              void* d_out, int out_size, void* d_ws, size_t ws_size,
                              hipStream_t stream) {
  (void)in_sizes; (void)n_in; (void)out_size; (void)ws_size;
  const int*   x_atom_type = (const int*)d_in[0];
  const float* x_feats     = (const float*)d_in[1];
  const int*   src         = (const int*)d_in[2];
  const int*   dst         = src + NEDGES;
  const int*   e_btype     = (const int*)d_in[3];
  const float* e_feats     = (const float*)d_in[4];
  const int*   batch       = (const int*)d_in[5];
  const float* dielec      = (const float*)d_in[6];
  const float* refract     = (const float*)d_in[7];
  auto F = [&](int i) { return (const float*)d_in[i]; };
  const float* atom_emb = F(8);
  const float* bond_emb = F(9);
  const float* dielec_b = F(10); const float* dielec_w = F(11);
  const float* k_b = F(12);      const float* k_w = F(13);
  // layer L: base = 14 + 14*L ; +0 e1b +1 e1w +2 e2b +3 e2w +4 e3b +5 e3w
  //          +6 m1b +7 m1w +8 m2b +9 m2w +10 u1b +11 u1w +12 u2b +13 u2w
  const float* la_b = F(70); const float* la_w = F(71);
  const float* lb_b = F(72); const float* lb_w = F(73);
  const float* o1_b = F(74); const float* o1_w = F(75);
  const float* o2_b = F(76); const float* o2_w = F(77);
  const float* rf_b = F(78); const float* rf_w = F(79);
  const float* seeds = F(80);
  const float* v_b = F(81);  const float* v_w = F(82);

  // ---- workspace carve-up (~270 MB)
  uintptr_t cur = (uintptr_t)d_ws;
  auto take = [&](size_t bytes) -> void* {
    void* p = (void*)cur; cur += (bytes + 255) & ~(size_t)255; return p;
  };
  _Float16* hbuf  = (_Float16*)take((size_t)NNODES * 128 * 2);
  _Float16* mebuf = (_Float16*)take((size_t)NEDGES * 128 * 2);
  float*    aggr  = (float*)take((size_t)NNODES * 128 * 4);
  _Float16* vbuf  = (_Float16*)take((size_t)NNODES * 128 * 2);
  float*    scores = (float*)take((size_t)NNODES * 16 * 4);
  unsigned* smax  = (unsigned*)take((size_t)BGRAPH * 16 * 4);
  float*    denom = (float*)take((size_t)BGRAPH * 16 * 4);
  float*    pooled = (float*)take((size_t)BGRAPH * 512 * 4);
  float*    esum  = (float*)take((size_t)BGRAPH * 128 * 4);
  float*    ecnt  = (float*)take((size_t)BGRAPH * 4);
  _Float16 *pm1[4], *pm2[4], *pu1[4], *pu2[4], *pe1[4], *pe2[4], *pe3[4];
  for (int L = 0; L < 4; ++L) {
    pm1[L] = (_Float16*)take(384 * 256 * 2); pm2[L] = (_Float16*)take(256 * 128 * 2);
    pu1[L] = (_Float16*)take(256 * 256 * 2); pu2[L] = (_Float16*)take(256 * 128 * 2);
    pe1[L] = (_Float16*)take(384 * 256 * 2); pe2[L] = (_Float16*)take(256 * 128 * 2);
    pe3[L] = (_Float16*)take(128 * 128 * 2);
  }
  _Float16* pkw = (_Float16*)take(128 * 128 * 2);
  _Float16* pvw = (_Float16*)take(128 * 128 * 2);
  _Float16* po1 = (_Float16*)take(704 * 256 * 2);
  _Float16* psB = (_Float16*)take(128 * 16 * 2);

  // ---- zero / init accumulators
  auto fill = [&](void* p, unsigned v, long n) {
    k_fill_u32<<<dim3((unsigned)((n + 255) / 256)), dim3(256), 0, stream>>>(
        (unsigned*)p, v, n);
  };
  fill(aggr, 0u, (long)NNODES * 128);
  fill(denom, 0u, BGRAPH * 16);
  fill(pooled, 0u, BGRAPH * 512);
  fill(esum, 0u, BGRAPH * 128);
  fill(ecnt, 0u, BGRAPH);
  fill(smax, 0x007FFFFFu, BGRAPH * 16);   // enc(-inf)

  // ---- pack weights into WMMA-B fragment layout (f16)
  auto pack = [&](const float* W, int K, int N, _Float16* dstp) {
    int total = K * N;
    k_pack_w<<<dim3((total + 255) / 256), dim3(256), 0, stream>>>(W, K, N, dstp);
  };
  for (int L = 0; L < 4; ++L) {
    int base = 14 + 14 * L;
    pack(F(base + 1), 384, 256, pe1[L]);
    pack(F(base + 3), 256, 128, pe2[L]);
    pack(F(base + 5), 128, 128, pe3[L]);
    pack(F(base + 7), 384, 256, pm1[L]);
    pack(F(base + 9), 256, 128, pm2[L]);
    pack(F(base + 11), 256, 256, pu1[L]);
    pack(F(base + 13), 256, 128, pu2[L]);
  }
  pack(k_w, 128, 128, pkw);
  pack(v_w, 128, 128, pvw);
  pack(o1_w, 704, 256, po1);
  k_pack_seeds<<<8, 256, 0, stream>>>(seeds, psB);

  // ---- input projections
  k_node_in<<<NNODES, 128, 0, stream>>>(x_atom_type, x_feats, atom_emb, la_w, la_b, hbuf);
  k_edge_in<<<NEDGES, 128, 0, stream>>>(e_btype, e_feats, bond_emb, lb_w, lb_b, mebuf);

  // ---- 4 message-passing layers
  const int etiles = NEDGES / 16;   // 37500
  const int ntilesN = NNODES / 16;  // 6250
  for (int L = 0; L < 4; ++L) {
    int base = 14 + 14 * L;
    k_edge_layer<<<etiles / EW, 64, 0, stream>>>(
        hbuf, mebuf, aggr, src, dst,
        pm1[L], F(base + 6), pm2[L], F(base + 8),
        pe1[L], F(base + 0), pe2[L], F(base + 2), pe3[L], F(base + 4), etiles);
    k_node_layer<<<ntilesN / EW, 64, 0, stream>>>(
        hbuf, aggr, pu1[L], F(base + 10), pu2[L], F(base + 12), ntilesN);
  }

  // ---- attention readout + pooling + head
  k_kv<<<(ntilesN + KW - 1) / KW, 128, 0, stream>>>(
      hbuf, pkw, k_b, pvw, v_b, psB, batch, scores, smax, vbuf, ntilesN);
  k_expsum<<<(NNODES * 16 + 255) / 256, 256, 0, stream>>>(scores, smax, denom, batch,
                                                          NNODES);
  k_pooled<<<NNODES, 128, 0, stream>>>(scores, denom, vbuf, batch, pooled);
  k_epool<<<NEDGES, 128, 0, stream>>>(mebuf, src, batch, esum, ecnt);
  k_final<<<BGRAPH / 16, 32, 0, stream>>>(pooled, esum, ecnt, dielec, refract,
                                          dielec_w, dielec_b, rf_w, rf_b,
                                          po1, o1_b, o2_w, o2_b, (float*)d_out);
}
